// Decoder_67645734913003
// MI455X (gfx1250) — matrix operand order repaired and checked by reading, not compiled, not run on hardware
//
#include <hip/hip_runtime.h>
#include <hip/hip_bf16.h>
#include <math.h>

// ---------------------------------------------------------------------------
// Problem constants
// ---------------------------------------------------------------------------
#define BB   128      // batch
#define SS   512      // encoder sequence length
#define HID  1024     // hidden
#define EMB  512      // embedding dim
#define VOUT 32000    // vocab / output

typedef __attribute__((ext_vector_type(16))) __bf16 v16bf;
typedef __attribute__((ext_vector_type(8)))  __bf16 v8bf;
typedef __attribute__((ext_vector_type(4)))  __bf16 v4bf;
typedef __attribute__((ext_vector_type(8)))  float  v8f;

static __device__ __forceinline__ v16bf cat8(v8bf lo, v8bf hi) {
    v16bf r;
#pragma unroll
    for (int i = 0; i < 8; ++i) { r[i] = lo[i]; r[i + 8] = hi[i]; }
    return r;
}

static __device__ __forceinline__ v4bf cvt4(float4 f) {
    v4bf o;
    o[0] = (__bf16)f.x; o[1] = (__bf16)f.y;
    o[2] = (__bf16)f.z; o[3] = (__bf16)f.w;
    return o;
}

// v_rcp_f32-based fast math: no IEEE divide expansion (v_div_scale/v_div_fmas)
static __device__ __forceinline__ float sigmoidf_(float x) {
    return __builtin_amdgcn_rcpf(1.0f + __expf(-x));
}

// Branch-free tanh: 1 - 2/(e^{2x}+1); exp->0 gives -1, exp->inf gives +1.
// Single v_exp_f32 + single v_rcp_f32, no EXEC-mask piecewise regions.
static __device__ __forceinline__ float tanh_fast(float x) {
    return 1.0f - 2.0f * __builtin_amdgcn_rcpf(__expf(2.0f * x) + 1.0f);
}

// ---------------------------------------------------------------------------
// 1) Embedding gather
// ---------------------------------------------------------------------------
__global__ void embed_kernel(const int* __restrict__ tok,
                             const float* __restrict__ table,
                             float* __restrict__ embedded) {
    int idx = blockIdx.x * blockDim.x + threadIdx.x;     // B*EMB
    int b = idx / EMB, e = idx % EMB;
    embedded[idx] = table[(size_t)tok[b] * EMB + e];
}

// ---------------------------------------------------------------------------
// 2) q = query @ Wa_w.T + Wa_b
// ---------------------------------------------------------------------------
__global__ void qproj_kernel(const float* __restrict__ query,
                             const float* __restrict__ Wa_w,
                             const float* __restrict__ Wa_b,
                             float* __restrict__ q) {
    int idx = blockIdx.x * blockDim.x + threadIdx.x;     // B*HID
    int b = idx >> 10, n = idx & (HID - 1);
    const float4* xq = (const float4*)(query + (size_t)b * HID);
    const float4* wr = (const float4*)(Wa_w + (size_t)n * HID);
    float acc = Wa_b[n];
#pragma unroll 4
    for (int k = 0; k < HID / 4; ++k) {
        float4 a = xq[k], w = wr[k];
        acc += a.x * w.x + a.y * w.y + a.z * w.z + a.w * w.w;
    }
    q[idx] = acc;
}

// ---------------------------------------------------------------------------
// 3) Fused attention scores (137 GFLOP GEMM + tanh/dot reduction, never
//    materializing k).  256 threads = 8 waves; 128 enc rows staged in LDS
//    (264 KB) + 16-row Ua_w K-panel (33 KB) -> 297 KB of the 320 KB WGP LDS.
//    Each wave owns a 16-row strip; v_wmma_f32_16x16x32_bf16 over K=1024.
// ---------------------------------------------------------------------------
#define LDA  (HID + 8)     // bf16 row stride: 2064 B (16B-aligned)
#define AROWS 128

__global__ __launch_bounds__(256, 1)
void attn_scores_kernel(const float* __restrict__ enc,
                        const float* __restrict__ Ua_w,
                        const float* __restrict__ Ua_b,
                        const float* __restrict__ Va_w,
                        const float* __restrict__ Va_b,
                        const float* __restrict__ q,
                        float* __restrict__ scores) {
    __shared__ __bf16 ldsA[AROWS * LDA];   // 128 enc rows  x K=1024
    __shared__ __bf16 ldsB[16 * LDA];      // 16 Ua_w rows  x K=1024

    const int tid  = threadIdx.x;
    const int wave = tid >> 5;
    const int lane = tid & 31;
    const int lm   = lane & 15;            // row (A) / col (B,C) index
    const int hi   = lane >> 4;            // lane half

    const int tilesPerB = SS / AROWS;                  // 4
    const int b  = blockIdx.x / tilesPerB;
    const int s0 = (blockIdx.x % tilesPerB) * AROWS;

    // Stage A tile (vectorized: float4 global load -> v4bf packed LDS store)
    const float* encBase = enc + ((size_t)b * SS + s0) * HID;
    for (int i4 = tid; i4 < AROWS * HID / 4; i4 += 256) {
        int e = i4 * 4, r = e >> 10, k = e & (HID - 1);
        *(v4bf*)&ldsA[r * LDA + k] =
            cvt4(*(const float4*)(encBase + (size_t)r * HID + k));
    }

    float acc[8];
#pragma unroll
    for (int r = 0; r < 8; ++r) acc[r] = 0.0f;

    const int rowA0 = wave * 16;

    for (int nc = 0; nc < HID / 16; ++nc) {            // 64 k-proj col chunks
        const int n0 = nc * 16;
        __syncthreads();
        for (int i4 = tid; i4 < 16 * HID / 4; i4 += 256) {
            int e = i4 * 4, r = e >> 10, k = e & (HID - 1);
            *(v4bf*)&ldsB[r * LDA + k] =
                cvt4(*(const float4*)(Ua_w + (size_t)(n0 + r) * HID + k));
        }
        __syncthreads();

        v8f c = {0.f, 0.f, 0.f, 0.f, 0.f, 0.f, 0.f, 0.f};
#pragma unroll 4
        for (int kc = 0; kc < HID / 32; ++kc) {
            const int k0 = kc * 32;
            const __bf16* pa = &ldsA[(rowA0 + lm) * LDA + k0 + hi * 8];
            v16bf a = cat8(*(const v8bf*)pa, *(const v8bf*)(pa + 16));
            const __bf16* pb = &ldsB[lm * LDA + k0 + hi * 8];
            v16bf bt = cat8(*(const v8bf*)pb, *(const v8bf*)(pb + 16));
            c = __builtin_amdgcn_wmma_f32_16x16x32_bf16(
                    false, a, false, bt, (short)0, c, false, false);
        }
        // Epilogue: lane holds column n = n0+lm of rows M = r + 8*hi
        const int n  = n0 + lm;
        const float qv = q[(size_t)b * HID + n] + Ua_b[n];
        const float vw = Va_w[n];
#pragma unroll
        for (int r = 0; r < 8; ++r)
            acc[r] += vw * tanh_fast(qv + c[r]);
    }

    // Reduce across the 16 lanes of each half (the 16 columns per chunk)
#pragma unroll
    for (int off = 8; off >= 1; off >>= 1) {
#pragma unroll
        for (int r = 0; r < 8; ++r)
            acc[r] += __shfl_xor(acc[r], off, 32);
    }
    if (lm == 0) {
        const float vb = Va_b[0];
#pragma unroll
        for (int r = 0; r < 8; ++r) {
            int row = rowA0 + r + 8 * hi;
            scores[(size_t)b * SS + s0 + row] = acc[r] + vb;
        }
    }
}

// ---------------------------------------------------------------------------
// 4) Row softmax over S=512
// ---------------------------------------------------------------------------
__global__ void softmax_kernel(const float* __restrict__ scores,
                               float* __restrict__ attn) {
    __shared__ float red[256];
    const int b = blockIdx.x, tid = threadIdx.x;
    const float x0 = scores[(size_t)b * SS + tid];
    const float x1 = scores[(size_t)b * SS + tid + 256];
    red[tid] = fmaxf(x0, x1);
    __syncthreads();
    for (int off = 128; off > 0; off >>= 1) {
        if (tid < off) red[tid] = fmaxf(red[tid], red[tid + off]);
        __syncthreads();
    }
    const float mx = red[0];
    __syncthreads();
    const float e0 = __expf(x0 - mx), e1 = __expf(x1 - mx);
    red[tid] = e0 + e1;
    __syncthreads();
    for (int off = 128; off > 0; off >>= 1) {
        if (tid < off) red[tid] += red[tid + off];
        __syncthreads();
    }
    const float inv = 1.0f / red[0];
    attn[(size_t)b * SS + tid]       = e0 * inv;
    attn[(size_t)b * SS + tid + 256] = e1 * inv;
}

// ---------------------------------------------------------------------------
// 5) context[b,h] = sum_s attn[b,s] * enc[b,s,h]
// ---------------------------------------------------------------------------
__global__ void context_kernel(const float* __restrict__ attn,
                               const float* __restrict__ enc,
                               float* __restrict__ ctx) {
    const int b = blockIdx.x;
    const int h = blockIdx.y * blockDim.x + threadIdx.x;
    const float* ep = enc + (size_t)b * SS * HID + h;
    const float* ap = attn + (size_t)b * SS;
    float acc = 0.0f;
#pragma unroll 4
    for (int s = 0; s < SS; ++s)
        acc += ap[s] * ep[(size_t)s * HID];
    ctx[(size_t)b * HID + h] = acc;
}

// ---------------------------------------------------------------------------
// 6) Fused GRU cell (torch gate order r,z,n): one thread per (b,j)
// ---------------------------------------------------------------------------
__global__ void gru_kernel(const float* __restrict__ embedded,
                           const float* __restrict__ ctx,
                           const float* __restrict__ query,
                           const float* __restrict__ w_ih,
                           const float* __restrict__ b_ih,
                           const float* __restrict__ w_hh,
                           const float* __restrict__ b_hh,
                           float* __restrict__ hnew) {
    const int idx = blockIdx.x * blockDim.x + threadIdx.x;  // B*HID
    const int b = idx >> 10, j = idx & (HID - 1);
    const int KI = EMB + HID;                               // 1536

    const float4* xe = (const float4*)(embedded + (size_t)b * EMB);
    const float4* xc = (const float4*)(ctx + (size_t)b * HID);
    const float4* wr = (const float4*)(w_ih + (size_t)j * KI);
    const float4* wz = (const float4*)(w_ih + (size_t)(HID + j) * KI);
    const float4* wn = (const float4*)(w_ih + (size_t)(2 * HID + j) * KI);

    float ir = b_ih[j], iz = b_ih[HID + j], in_ = b_ih[2 * HID + j];
    for (int k = 0; k < EMB / 4; ++k) {
        float4 x = xe[k], a = wr[k], c = wz[k], d = wn[k];
        ir  += a.x * x.x + a.y * x.y + a.z * x.z + a.w * x.w;
        iz  += c.x * x.x + c.y * x.y + c.z * x.z + c.w * x.w;
        in_ += d.x * x.x + d.y * x.y + d.z * x.z + d.w * x.w;
    }
    const int o = EMB / 4;
    for (int k = 0; k < HID / 4; ++k) {
        float4 x = xc[k], a = wr[o + k], c = wz[o + k], d = wn[o + k];
        ir  += a.x * x.x + a.y * x.y + a.z * x.z + a.w * x.w;
        iz  += c.x * x.x + c.y * x.y + c.z * x.z + c.w * x.w;
        in_ += d.x * x.x + d.y * x.y + d.z * x.z + d.w * x.w;
    }

    const float4* hq = (const float4*)(query + (size_t)b * HID);
    const float4* vr = (const float4*)(w_hh + (size_t)j * HID);
    const float4* vz = (const float4*)(w_hh + (size_t)(HID + j) * HID);
    const float4* vn = (const float4*)(w_hh + (size_t)(2 * HID + j) * HID);
    float hr = b_hh[j], hz = b_hh[HID + j], hn = b_hh[2 * HID + j];
    for (int k = 0; k < HID / 4; ++k) {
        float4 x = hq[k], a = vr[k], c = vz[k], d = vn[k];
        hr += a.x * x.x + a.y * x.y + a.z * x.z + a.w * x.w;
        hz += c.x * x.x + c.y * x.y + c.z * x.z + c.w * x.w;
        hn += d.x * x.x + d.y * x.y + d.z * x.z + d.w * x.w;
    }

    const float r = sigmoidf_(ir + hr);
    const float z = sigmoidf_(iz + hz);
    const float n = tanh_fast(in_ + r * hn);
    hnew[idx] = (1.0f - z) * n + z * query[(size_t)b * HID + j];
}

// ---------------------------------------------------------------------------
// 7) prediction = h_new @ fc_w.T + fc_b  via WMMA bf16.
//    All 128 h_new rows staged in LDS (264 KB); 8 waves each own a 16-col
//    tile; per K-chunk one in-register fc_w fragment feeds 8 WMMAs
//    (8 m-tiles, 8 accumulators) -> fc_w streamed from HBM exactly once.
// ---------------------------------------------------------------------------
__global__ __launch_bounds__(256, 1)
void fc_kernel(const float* __restrict__ hnew,
               const float* __restrict__ fc_w,
               const float* __restrict__ fc_b,
               float* __restrict__ pred) {
    __shared__ __bf16 ldsA[BB * LDA];      // all of h_new, bf16

    const int tid  = threadIdx.x;
    const int wave = tid >> 5;
    const int lane = tid & 31;
    const int lm   = lane & 15;
    const int hi   = lane >> 4;

    const int n0 = (blockIdx.x * 8 + wave) * 16;       // this wave's 16 cols

    for (int i4 = tid; i4 < BB * HID / 4; i4 += 256) {
        int e = i4 * 4, r = e >> 10, k = e & (HID - 1);
        *(v4bf*)&ldsA[r * LDA + k] =
            cvt4(*(const float4*)(hnew + (size_t)r * HID + k));
    }
    __syncthreads();

    v8f c[8];
#pragma unroll
    for (int mt = 0; mt < 8; ++mt)
        c[mt] = (v8f){0.f, 0.f, 0.f, 0.f, 0.f, 0.f, 0.f, 0.f};

    const float* wrow = fc_w + (size_t)(n0 + lm) * HID;   // this lane's column
    for (int kc = 0; kc < HID / 32; ++kc) {
        const int k0 = kc * 32;
        if (k0 + 256 < HID)
            __builtin_prefetch(wrow + k0 + 256, 0, 0);    // global_prefetch_b8
        const float4* p4 = (const float4*)(wrow + k0 + hi * 8);
        v16bf bt;
#pragma unroll
        for (int jj = 0; jj < 4; ++jj) {
            float4 f = p4[(jj & 1) + (jj >> 1) * 4];
            bt[4 * jj + 0] = (__bf16)f.x;
            bt[4 * jj + 1] = (__bf16)f.y;
            bt[4 * jj + 2] = (__bf16)f.z;
            bt[4 * jj + 3] = (__bf16)f.w;
        }
#pragma unroll
        for (int mt = 0; mt < 8; ++mt) {
            const __bf16* pa = &ldsA[(mt * 16 + lm) * LDA + k0 + hi * 8];
            v16bf a = cat8(*(const v8bf*)pa, *(const v8bf*)(pa + 16));
            c[mt] = __builtin_amdgcn_wmma_f32_16x16x32_bf16(
                        false, a, false, bt, (short)0, c[mt], false, false);
        }
    }

    const int n = n0 + lm;
    const float bias = fc_b[n];
#pragma unroll
    for (int mt = 0; mt < 8; ++mt) {
#pragma unroll
        for (int r = 0; r < 8; ++r) {
            int row = mt * 16 + r + 8 * hi;
            pred[(size_t)row * VOUT + n] = c[mt][r] + bias;
        }
    }
}

// ---------------------------------------------------------------------------
// Launcher
// ---------------------------------------------------------------------------
extern "C" void kernel_launch(void* const* d_in, const int* in_sizes, int n_in,
                              void* d_out, int out_size, void* d_ws, size_t ws_size,
                              hipStream_t stream) {
    const int*   tok    = (const int*)  d_in[0];
    const float* query  = (const float*)d_in[1];   // (1,B,H) == (B,H)
    const float* enc    = (const float*)d_in[2];
    const float* table  = (const float*)d_in[3];
    const float* Wa_w   = (const float*)d_in[4];
    const float* Wa_b   = (const float*)d_in[5];
    const float* Ua_w   = (const float*)d_in[6];
    const float* Ua_b   = (const float*)d_in[7];
    const float* Va_w   = (const float*)d_in[8];
    const float* Va_b   = (const float*)d_in[9];
    const float* w_ih   = (const float*)d_in[10];
    const float* b_ih   = (const float*)d_in[11];
    const float* w_hh   = (const float*)d_in[12];
    const float* b_hh   = (const float*)d_in[13];
    const float* fc_w   = (const float*)d_in[14];
    const float* fc_b   = (const float*)d_in[15];

    float* ws       = (float*)d_ws;
    float* q        = ws;                          // B*H
    float* embedded = ws + BB * HID;               // B*E
    float* ctx      = embedded + BB * EMB;         // B*H
    float* scores   = ctx + BB * HID;              // B*S

    float* pred     = (float*)d_out;               // B*VOUT
    float* hnew_out = pred + (size_t)BB * VOUT;    // B*H
    float* attn_out = hnew_out + BB * HID;         // B*S

    embed_kernel      <<<BB * EMB / 256, 256, 0, stream>>>(tok, table, embedded);
    qproj_kernel      <<<BB * HID / 256, 256, 0, stream>>>(query, Wa_w, Wa_b, q);
    attn_scores_kernel<<<BB * SS / AROWS, 256, 0, stream>>>(enc, Ua_w, Ua_b, Va_w,
                                                            Va_b, q, scores);
    softmax_kernel    <<<BB,             256, 0, stream>>>(scores, attn_out);
    context_kernel    <<<dim3(BB, HID / 256), 256, 0, stream>>>(attn_out, enc, ctx);
    gru_kernel        <<<BB * HID / 256, 256, 0, stream>>>(embedded, ctx, query,
                                                           w_ih, b_ih, w_hh, b_hh,
                                                           hnew_out);
    fc_kernel         <<<VOUT / 128,     256, 0, stream>>>(hnew_out, fc_w,
                                                           fc_b, pred);
}
